// ST_HMR_78486232367524
// MI455X (gfx1250) — matrix operand order repaired and checked by reading, not compiled, not run on hardware
//
#include <hip/hip_runtime.h>
#include <hip/hip_bf16.h>

typedef __attribute__((ext_vector_type(16))) __bf16 v16bf;
typedef __attribute__((ext_vector_type(8)))  float  v8f;

#define BB    8
#define TT    32
#define NBONE 24
#define HH    256
#define STOT  (BB*TT*NBONE*HH)   // 1572864 elements per output tensor

// activation index [b][t][n][k] in a [B,T,N,H] tensor
__device__ __forceinline__ size_t aidx(int b, int t, int n, int k) {
    return (((size_t)b * TT + t) * NBONE + n) * HH + k;
}

__device__ __forceinline__ unsigned pack_bf16(float a, float b) {
    union { __bf16 h[2]; unsigned u; } r;
    r.h[0] = (__bf16)a; r.h[1] = (__bf16)b;
    return r.u;
}

__device__ __forceinline__ float sigf(float x) { return 1.f / (1.f + __expf(-x)); }

union FragAB { v16bf v; unsigned u[8]; uint4 q[2]; };

// ---------------------------------------------------------------------------
// Kernel 1: all 8 gate GEMMs fused + gating epilogue.
// Per (g,t): C[192x256] = A_cat[192x1536] @ W_cat[1536x256]  (bf16 WMMA, f32 acc)
// Block: 128 threads (4 waves). Tile: M=64 (wave m-subtiles), N=16.
// Grid: x = 3 Mtiles * 16 Ntiles = 48, y = T = 32.
// LDS double-buffered; all 8 B fragments preloaded (b128) so WMMAs issue
// back-to-back behind a single dscnt wait.
// ---------------------------------------------------------------------------
__global__ __launch_bounds__(128)
void st_hmr_fused1(const float* __restrict__ h,   const float* __restrict__ c_h,
                   const float* __restrict__ p,   const float* __restrict__ gt,
                   const float* __restrict__ cgt,
                   const float* __restrict__ U,   const float* __restrict__ Wt,
                   const float* __restrict__ Ws,  const float* __restrict__ Zt,
                   const float* __restrict__ bias, float* __restrict__ out) {
    // A: 64 rows x 16 k-pairs, padded 16->20 dwords (b128-aligned, conflict-free)
    __shared__ unsigned Alds[2][64][20];
    // B: 8 gates x [16 n][16 k-pairs], padded 16->20 dwords (b128-aligned)
    __shared__ unsigned Blds[2][8][16][20];

    const int t    = blockIdx.y;
    const int mt   = (int)blockIdx.x / 16;          // 0..2
    const int nt   = (int)blockIdx.x % 16;          // 0..15
    const int m0   = mt * 64, n0 = nt * 16;
    const int tid  = threadIdx.x;
    const int lane = tid & 31, wave = tid >> 5;
    const int hi   = lane >> 4, lr = lane & 15;

    const size_t wbH  = (size_t)HH * HH;
    const size_t wb3H = (size_t)3 * HH * HH;

    // per-thread fixed staging coordinates
    const int mA  = tid >> 1;                       // A row within tile (0..63)
    const int khA = tid & 1;                        // A k-half (0 or 16 floats)
    const int bA  = (m0 + mA) / NBONE;
    const int nbA = (m0 + mA) % NBONE;
    const int klB = tid >> 2;                       // B k row (0..31)
    const int nqB = tid & 3;                        // B n quad (0..3)

    v8f acc[8] = {};

    auto stage = [&](int kt, int pb) {
        const int seg = kt >> 3;                    // 0..5 (256-wide K segment)
        const int ko  = (kt & 7) * 32;

        // ---- A source selection (uniform) ----
        const float* ab = p; int toff = t; bool tv = true; int noff = 0;
        switch (seg) {
            case 0: ab = p;  break;
            case 1: ab = h;  toff = t - 1; tv = (t > 0);      break;
            case 2: ab = h;  break;
            case 3: ab = h;  toff = t + 1; tv = (t < TT - 1); break;
            case 4: ab = h;  noff = -1;    break;             // h_s_before
            default: ab = gt; break;
        }
        const int nb2 = nbA + noff;
        float4 f0, f1, f2, f3;
        if (tv && nb2 >= 0) {
            const float4* src = (const float4*)(ab + aidx(bA, toff, nb2, ko + khA * 16));
            f0 = src[0]; f1 = src[1]; f2 = src[2]; f3 = src[3];
        } else {
            f0 = f1 = f2 = f3 = make_float4(0.f, 0.f, 0.f, 0.f);
        }
        uint4* adst = (uint4*)&Alds[pb][mA][khA * 8];
        adst[0] = make_uint4(pack_bf16(f0.x, f0.y), pack_bf16(f0.z, f0.w),
                             pack_bf16(f1.x, f1.y), pack_bf16(f1.z, f1.w));
        adst[1] = make_uint4(pack_bf16(f2.x, f2.y), pack_bf16(f2.z, f2.w),
                             pack_bf16(f3.x, f3.y), pack_bf16(f3.z, f3.w));

        // ---- B: one float4 (4 n) per gate per thread, transposed bf16 store ----
        const float* wbp[8];
        #pragma unroll
        for (int gg = 0; gg < 8; ++gg) {
            size_t gtoff = (size_t)gg * TT + t;
            const float* base;
            if (seg == 0)      base = U  + gtoff * wbH;
            else if (seg <= 3) base = Wt + gtoff * wb3H + (size_t)(seg - 1) * wbH;
            else if (seg == 4) base = Ws + gtoff * wbH;
            else               base = Zt + gtoff * wbH;
            wbp[gg] = base + (size_t)(ko + klB) * HH + n0 + nqB * 4;
        }
        #pragma unroll
        for (int gg = 0; gg < 8; ++gg) {
            float4 w = *(const float4*)wbp[gg];
            __bf16* br = (__bf16*)&Blds[pb][gg][0][0];
            br[(nqB * 4 + 0) * 40 + klB] = (__bf16)w.x;
            br[(nqB * 4 + 1) * 40 + klB] = (__bf16)w.y;
            br[(nqB * 4 + 2) * 40 + klB] = (__bf16)w.z;
            br[(nqB * 4 + 3) * 40 + klB] = (__bf16)w.w;
        }
        // prefetch the following K-slice of this segment's weights
        if ((kt & 7) != 7)
            __builtin_prefetch(wbp[tid & 7] + (size_t)32 * HH, 0, 1);
    };

    auto compute = [&](int pb) {
        FragAB fa;
        fa.q[0] = *(const uint4*)&Alds[pb][wave * 16 + lr][hi * 4];
        fa.q[1] = *(const uint4*)&Alds[pb][wave * 16 + lr][8 + hi * 4];

        FragAB fb[8];
        #pragma unroll
        for (int gg = 0; gg < 8; ++gg) {
            fb[gg].q[0] = *(const uint4*)&Blds[pb][gg][lr][hi * 4];
            fb[gg].q[1] = *(const uint4*)&Blds[pb][gg][lr][8 + hi * 4];
        }
        #pragma unroll
        for (int gg = 0; gg < 8; ++gg)
            acc[gg] = __builtin_amdgcn_wmma_f32_16x16x32_bf16(
                false, fa.v, false, fb[gg].v, (short)0, acc[gg], false, false);
    };

    stage(0, 0);
    for (int kt = 0; kt < 48; ++kt) {
        __syncthreads();
        if (kt + 1 < 48) stage(kt + 1, (kt + 1) & 1);
        compute(kt & 1);
    }

    // ---- fused gating epilogue: lane owns col = n0+lr, rows v + 8*hi ----
    const int col = n0 + lr;
    #pragma unroll
    for (int v8i = 0; v8i < 8; ++v8i) {
        int mrow = m0 + wave * 16 + v8i + (hi << 3);
        int b = mrow / NBONE, nb = mrow % NBONE;
        size_t ai = aidx(b, t, nb, col);

        float pr[8];
        #pragma unroll
        for (int gg = 0; gg < 8; ++gg)
            pr[gg] = acc[gg][v8i] + bias[((size_t)gg * TT + t) * HH + col];

        float i_n  = sigf(pr[0]);
        float f_lt = sigf(pr[1]);
        float f_ft = sigf(pr[2]);
        float f_rt = sigf(pr[3]);
        float f_s  = sigf(pr[4]);
        float f_gt = sigf(pr[5]);
        float o_n  = sigf(pr[6]);
        float c_n  = tanhf(pr[7]);

        float ctb = (t > 0)      ? c_h[ai - (size_t)NBONE * HH] : 0.f;
        float cta = (t < TT - 1) ? c_h[ai + (size_t)NBONE * HH] : 0.f;
        float csb = (nb > 0)     ? c_h[ai - HH] : 0.f;

        float chn = c_n * i_n + ctb * f_lt + c_h[ai] * f_ft
                  + cta * f_rt + csb * f_s + cgt[ai] * f_gt;
        float hn  = o_n * tanhf(chn);

        out[ai]                = hn;    // h_n
        out[(size_t)STOT + ai] = chn;   // c_h_n
    }
}

// ---------------------------------------------------------------------------
// Kernel 2: g_t update. gpre[g2,t] = [h_n ; g_t] @ [Wg ; Zg], K=512, 3 gates,
// fused LSTM-style epilogue -> g_t_n, c_g_t_n. Same pipelined structure.
// ---------------------------------------------------------------------------
__global__ __launch_bounds__(128)
void st_hmr_fused2(const float* __restrict__ hn,  const float* __restrict__ gt,
                   const float* __restrict__ cgt,
                   const float* __restrict__ Wg,  const float* __restrict__ Zg,
                   const float* __restrict__ bg,  float* __restrict__ out) {
    __shared__ unsigned Alds[2][64][20];
    __shared__ unsigned Blds[2][3][16][20];

    const int t    = blockIdx.y;
    const int mt   = (int)blockIdx.x / 16;
    const int nt   = (int)blockIdx.x % 16;
    const int m0   = mt * 64, n0 = nt * 16;
    const int tid  = threadIdx.x;
    const int lane = tid & 31, wave = tid >> 5;
    const int hi   = lane >> 4, lr = lane & 15;

    const size_t wbH = (size_t)HH * HH;

    const int mA  = tid >> 1;
    const int khA = tid & 1;
    const int bA  = (m0 + mA) / NBONE;
    const int nbA = (m0 + mA) % NBONE;
    const int klB = tid >> 2;
    const int nqB = tid & 3;

    v8f acc[3] = {};

    auto stage = [&](int kt, int pb) {
        const int seg = kt >> 3;                    // 0: h_n, 1: g_t
        const int ko  = (kt & 7) * 32;
        const float* ab = seg ? gt : hn;

        const float4* src = (const float4*)(ab + aidx(bA, t, nbA, ko + khA * 16));
        float4 f0 = src[0], f1 = src[1], f2 = src[2], f3 = src[3];
        uint4* adst = (uint4*)&Alds[pb][mA][khA * 8];
        adst[0] = make_uint4(pack_bf16(f0.x, f0.y), pack_bf16(f0.z, f0.w),
                             pack_bf16(f1.x, f1.y), pack_bf16(f1.z, f1.w));
        adst[1] = make_uint4(pack_bf16(f2.x, f2.y), pack_bf16(f2.z, f2.w),
                             pack_bf16(f3.x, f3.y), pack_bf16(f3.z, f3.w));

        const float* wsel = seg ? Zg : Wg;
        #pragma unroll
        for (int gg = 0; gg < 3; ++gg) {
            const float* wbp = wsel + ((size_t)gg * TT + t) * wbH
                             + (size_t)(ko + klB) * HH + n0 + nqB * 4;
            float4 w = *(const float4*)wbp;
            __bf16* br = (__bf16*)&Blds[pb][gg][0][0];
            br[(nqB * 4 + 0) * 40 + klB] = (__bf16)w.x;
            br[(nqB * 4 + 1) * 40 + klB] = (__bf16)w.y;
            br[(nqB * 4 + 2) * 40 + klB] = (__bf16)w.z;
            br[(nqB * 4 + 3) * 40 + klB] = (__bf16)w.w;
        }
        if ((kt & 7) != 7) {
            const float* pf = wsel + ((size_t)(tid % 3) * TT + t) * wbH
                            + (size_t)(ko + 32 + klB) * HH + n0;
            __builtin_prefetch(pf, 0, 1);
        }
    };

    auto compute = [&](int pb) {
        FragAB fa;
        fa.q[0] = *(const uint4*)&Alds[pb][wave * 16 + lr][hi * 4];
        fa.q[1] = *(const uint4*)&Alds[pb][wave * 16 + lr][8 + hi * 4];

        FragAB fb[3];
        #pragma unroll
        for (int gg = 0; gg < 3; ++gg) {
            fb[gg].q[0] = *(const uint4*)&Blds[pb][gg][lr][hi * 4];
            fb[gg].q[1] = *(const uint4*)&Blds[pb][gg][lr][8 + hi * 4];
        }
        #pragma unroll
        for (int gg = 0; gg < 3; ++gg)
            acc[gg] = __builtin_amdgcn_wmma_f32_16x16x32_bf16(
                false, fa.v, false, fb[gg].v, (short)0, acc[gg], false, false);
    };

    stage(0, 0);
    for (int kt = 0; kt < 16; ++kt) {
        __syncthreads();
        if (kt + 1 < 16) stage(kt + 1, (kt + 1) & 1);
        compute(kt & 1);
    }

    const int col = n0 + lr;
    #pragma unroll
    for (int v8i = 0; v8i < 8; ++v8i) {
        int mrow = m0 + wave * 16 + v8i + (hi << 3);
        int b = mrow / NBONE, nb = mrow % NBONE;
        size_t ai = aidx(b, t, nb, col);

        float fg = sigf(acc[0][v8i] + bg[((size_t)0 * TT + t) * HH + col]);
        float gc = tanhf(acc[1][v8i] + bg[((size_t)1 * TT + t) * HH + col]);
        float og = sigf(acc[2][v8i] + bg[((size_t)2 * TT + t) * HH + col]);

        float cg  = cgt[ai];
        float cgn = fg * cg + (1.f - fg) * gc;
        float gtn = og * tanhf(cgn);

        out[(size_t)2 * STOT + ai] = gtn;   // g_t_n
        out[(size_t)3 * STOT + ai] = cgn;   // c_g_t_n
    }
}

// ---------------------------------------------------------------------------
extern "C" void kernel_launch(void* const* d_in, const int* in_sizes, int n_in,
                              void* d_out, int out_size, void* d_ws, size_t ws_size,
                              hipStream_t stream) {
    (void)in_sizes; (void)n_in; (void)out_size; (void)d_ws; (void)ws_size;
    const float* h    = (const float*)d_in[0];
    const float* c_h  = (const float*)d_in[1];
    const float* p    = (const float*)d_in[2];
    const float* g_t  = (const float*)d_in[3];
    const float* c_gt = (const float*)d_in[4];
    const float* U    = (const float*)d_in[5];
    const float* Wt   = (const float*)d_in[6];
    const float* Ws   = (const float*)d_in[7];
    const float* Zt   = (const float*)d_in[8];
    const float* b    = (const float*)d_in[9];
    const float* Wg   = (const float*)d_in[10];
    const float* Zg   = (const float*)d_in[11];
    const float* bg   = (const float*)d_in[12];
    float* out = (float*)d_out;

    dim3 grid(48, TT, 1);   // 3 Mtiles x 16 Ntiles, one y per timestep
    st_hmr_fused1<<<grid, 128, 0, stream>>>(h, c_h, p, g_t, c_gt,
                                            U, Wt, Ws, Zt, b, out);
    // h_n now lives at out[0 .. STOT)
    st_hmr_fused2<<<grid, 128, 0, stream>>>(out, g_t, c_gt, Wg, Zg, bg, out);
}
